// DifferentiableDAG_53480932770071
// MI455X (gfx1250) — matrix operand order repaired and checked by reading, not compiled, not run on hardware
//
#include <hip/hip_runtime.h>

#define BB 4
#define NN 8
#define TT 64
#define HH 512
#define NTOT 512          // N*T
#define NOPS9 9
#define LDB 512           // row stride of every B operand in this pipeline

typedef __attribute__((ext_vector_type(16))) _Float16 v16h;
typedef __attribute__((ext_vector_type(8)))  float    v8f;

// ---------------------------------------------------------------------------
// Deterministic gumbel noise (hash-based counter RNG)
// ---------------------------------------------------------------------------
__device__ __forceinline__ float gumbel_noise(unsigned seed, unsigned x,
                                              unsigned y, unsigned z) {
  unsigned h = seed * 0x9E3779B9u;
  h ^= x + 0x9E3779B9u + (h << 6) + (h >> 2);
  h ^= y + 0x9E3779B9u + (h << 6) + (h >> 2);
  h ^= z + 0x9E3779B9u + (h << 6) + (h >> 2);
  h ^= h >> 16; h *= 0x7FEB352Du; h ^= h >> 15; h *= 0x846CA68Bu; h ^= h >> 16;
  float u = (float)(h >> 8) * (1.0f / 16777216.0f) + 1e-7f;
  return -__logf(-__logf(u));
}

// ---------------------------------------------------------------------------
// f32-in/f32-out GEMM on v_wmma_f32_16x16x32_f16, wave tile 32(M) x 64(N).
// C[M,N] = A[M,K] * B[K,N] (+bias).  TRANSB==1 => B given as Bt[N,K] row-major.
// B row stride is the compile-time constant LDB so all fragment elements are
// immediate-offset loads off one base pointer per operand.
// Fragment layouts used by this kernel (wave32; same k order in A and B):
//   A 16x32 f16 : lane<16 row=lane, elems 0..7 = K0..7,  8..15 = K8..15
//                 lane>=16 row=lane-16, elems = K16..23 / K24..31
//   B 32x16 f16 : lane<16 col=lane,  elems j = K j ; lane>=16 col=lane-16, K 16+j
//   C 16x16 f32 : vgpr r -> row r (+8 for upper half-wave), col = lane%16
// ---------------------------------------------------------------------------
template <int TRANSB>
__global__ __launch_bounds__(128) void wmma_gemm_kernel(
    const float* __restrict__ A, const float* __restrict__ Bm,
    const float* __restrict__ bias, float* __restrict__ C,
    int M, int N, int K, int lda, int ldc,
    long sA, long sB, long sC) {
  int batch = blockIdx.y;
  A += sA * batch; Bm += sB * batch; C += sC * batch;
  int wave = blockIdx.x * (blockDim.x >> 5) + (threadIdx.x >> 5);
  int lane = threadIdx.x & 31;
  int ntiles = N >> 6;
  int tiles = (M >> 5) * ntiles;
  if (wave >= tiles) return;          // wave-uniform exit (EXEC stays all-1s)
  int m0 = (wave / ntiles) << 5;
  int n0 = (wave % ntiles) << 6;
  int r  = lane & 15;
  int hi = lane >> 4;

  v8f acc[2][4] = {};
  const float* Arow0 = A + (long)(m0 + r) * lda + hi * 16;
  const float* Arow1 = A + (long)(m0 + 16 + r) * lda + hi * 16;
  // single B base pointer; all fragment elements at immediate offsets
  const float* Bbase = TRANSB ? (Bm + (long)(n0 + r) * LDB + hi * 16)
                              : (Bm + (long)(hi * 16) * LDB + n0 + r);

  for (int k0 = 0; k0 < K; k0 += 32) {
    v16h a0, a1;
#pragma unroll
    for (int j = 0; j < 8; ++j) {
      a0[j]     = (_Float16)Arow0[j];        // K: hi*16+j
      a0[j + 8] = (_Float16)Arow0[8 + j];    // K: hi*16+8+j
      a1[j]     = (_Float16)Arow1[j];
      a1[j + 8] = (_Float16)Arow1[8 + j];
    }
    v16h bf[4];
#pragma unroll
    for (int s = 0; s < 4; ++s) {
#pragma unroll
      for (int j = 0; j < 16; ++j) {
        bf[s][j] = TRANSB ? (_Float16)Bbase[s * 16 * LDB + j]
                          : (_Float16)Bbase[j * LDB + s * 16];
      }
    }
    if (k0 + 32 < K) {  // prefetch next k-slab (global_prefetch_b8)
      __builtin_prefetch(Arow0 + 32, 0, 0);
      __builtin_prefetch(Arow1 + 32, 0, 0);
      __builtin_prefetch(TRANSB ? (Bbase + 32) : (Bbase + 32 * LDB), 0, 0);
    }
#pragma unroll
    for (int s = 0; s < 4; ++s) {
      acc[0][s] = __builtin_amdgcn_wmma_f32_16x16x32_f16(
          false, a0, false, bf[s], (short)0, acc[0][s], false, false);
      acc[1][s] = __builtin_amdgcn_wmma_f32_16x16x32_f16(
          false, a1, false, bf[s], (short)0, acc[1][s], false, false);
    }
    Arow0 += 32; Arow1 += 32;
    Bbase += TRANSB ? 32 : 32 * LDB;
  }

#pragma unroll
  for (int mi = 0; mi < 2; ++mi) {
#pragma unroll
    for (int s = 0; s < 4; ++s) {
      int col = n0 + s * 16 + r;
      float bv = bias ? bias[col] : 0.0f;
      float* cp = C + (long)(m0 + mi * 16 + hi * 8) * ldc + col;
#pragma unroll
      for (int rr = 0; rr < 8; ++rr) cp[(long)rr * ldc] = acc[mi][s][rr] + bv;
    }
  }
}

// ---------------------------------------------------------------------------
// Elementwise prep: opnd = operand_ctx + step0 ; opc = op_ctx + step0
// ---------------------------------------------------------------------------
__global__ void addstep_kernel(const float* __restrict__ octx,
                               const float* __restrict__ opctx,
                               const float* __restrict__ step0,
                               float* __restrict__ OPND,
                               float* __restrict__ OPC) {
  int i = blockIdx.x * blockDim.x + threadIdx.x;
  if (i >= BB * TT * HH) return;
  float s = step0[i & (HH - 1)];
  OPND[i] = octx[i] + s;
  OPC[i]  = opctx[i] + s;
}

// ---------------------------------------------------------------------------
// Precompute rank-1 MLP folds:
//   wse3[h] = sum_k Wse[k]*Wmlp[1024+k][h],  wse4 with 1536 block,
//   cc[h]   = sum_k bse[k]*(Wmlp[1024+k][h]+Wmlp[1536+k][h]) + bmlp[h]
// ---------------------------------------------------------------------------
__global__ void prep_kernel(const float* __restrict__ Wse,
                            const float* __restrict__ bse,
                            const float* __restrict__ Wmlp,
                            const float* __restrict__ bmlp,
                            float* __restrict__ wse3, float* __restrict__ wse4,
                            float* __restrict__ cc) {
  int h = blockIdx.x * blockDim.x + threadIdx.x;
  if (h >= HH) return;
  float s3 = 0.f, s4 = 0.f, c = 0.f;
  for (int k = 0; k < HH; ++k) {
    float w3 = Wmlp[(1024 + k) * HH + h];
    float w4 = Wmlp[(1536 + k) * HH + h];
    s3 += Wse[k] * w3;
    s4 += Wse[k] * w4;
    c  += bse[k] * (w3 + w4);
  }
  wse3[h] = s3; wse4[h] = s4; cc[h] = c + bmlp[h];
}

// ---------------------------------------------------------------------------
// Pass-through copy of node_embeds / node_values into the (N+1)-slot outputs
// ---------------------------------------------------------------------------
__global__ void copy_kernel(const float* __restrict__ emb,
                            const float* __restrict__ vals,
                            float* __restrict__ out) {
  int i = blockIdx.x * blockDim.x + threadIdx.x;
  const int NE = BB * NN * TT * HH;          // 1048576
  if (i < NE) {
    int b = i / (NN * TT * HH);
    out[i + b * TT * HH] = emb[i];           // out stride has one extra n-slot
  } else {
    int j = i - NE;
    if (j < BB * NN * TT) {
      int b = j / (NN * TT);
      out[(size_t)BB * (NN + 1) * TT * HH + j + b * TT] = vals[j];
    }
  }
}

// ---------------------------------------------------------------------------
// Block reductions (512 threads)
// ---------------------------------------------------------------------------
__device__ __forceinline__ float block_reduce_max(float v, float* red) {
  int tid = threadIdx.x;
  red[tid] = v; __syncthreads();
#pragma unroll
  for (int off = 256; off >= 1; off >>= 1) {
    if (tid < off) red[tid] = fmaxf(red[tid], red[tid + off]);
    __syncthreads();
  }
  float r = red[0]; __syncthreads();
  return r;
}
__device__ __forceinline__ float block_reduce_sum(float v, float* red) {
  int tid = threadIdx.x;
  red[tid] = v; __syncthreads();
#pragma unroll
  for (int off = 256; off >= 1; off >>= 1) {
    if (tid < off) red[tid] += red[tid + off];
    __syncthreads();
  }
  float r = red[0]; __syncthreads();
  return r;
}

// ---------------------------------------------------------------------------
// Attention + op-selection kernel.  One 512-thread block per (b,t).
//  - gathers scores from S1/S2 via l->(n,tau) map, mask/clamp/valid
//  - gumbel softmax over l (block reductions)
//  - scatters att back to the flat (n*T+tau) grid -> A1/A2 (GEMM-ready)
//  - v1/v2 reductions, opq@Wsel logits, hard gumbel op pick, new_val
// ---------------------------------------------------------------------------
__global__ __launch_bounds__(512) void attn_kernel(
    const float* __restrict__ S1, const float* __restrict__ S2,
    const float* __restrict__ OPQ, const float* __restrict__ Wsel,
    const float* __restrict__ bsel, const float* __restrict__ nv,
    float* __restrict__ A1, float* __restrict__ A2,
    float* __restrict__ V1, float* __restrict__ V2,
    float* __restrict__ OPW, float* __restrict__ outv) {
  __shared__ float a1s[512], a2s[512], red[512];
  int bt = blockIdx.x, b = bt >> 6, t = bt & 63;
  int l = threadIdx.x;
  int tp1 = t + 1;
  bool valid = l < NN * tp1;

  float y1 = -__builtin_inff(), y2 = -__builtin_inff();
  if (valid) {
    int n = l / tp1, tau = l - n * tp1;
    int nt = (n << 6) + tau;
    float x1 = S1[(bt << 9) + nt];
    float x2 = S2[(bt << 9) + nt];
    bool m = (l & 63) <= t;
    if (!m) { x1 = -1e9f; x2 = -1e9f; }
    x1 = fminf(fmaxf(x1, -40.f), 40.f);
    x2 = fminf(fmaxf(x2, -40.f), 40.f);
    y1 = (x1 + gumbel_noise(1u, b, t, l)) * 0.5f;   // /TAU, TAU=2
    y2 = (x2 + gumbel_noise(2u, b, t, l)) * 0.5f;
  }
  float m1 = block_reduce_max(y1, red);
  float e1 = valid ? __expf(y1 - m1) : 0.f;
  float d1 = block_reduce_sum(e1, red);
  float att1 = valid ? (e1 / d1 + 1e-10f) : 0.f;
  float m2 = block_reduce_max(y2, red);
  float e2 = valid ? __expf(y2 - m2) : 0.f;
  float d2 = block_reduce_sum(e2, red);
  float att2 = valid ? (e2 / d2 + 1e-10f) : 0.f;

  a1s[l] = att1; a2s[l] = att2; __syncthreads();

  // scatter to (n*T + tau) layout: valid target iff tau<=t, src l = n*(t+1)+tau
  int tau2 = l & 63, n2 = l >> 6;
  float o1 = 0.f, o2 = 0.f;
  if (tau2 <= t) {
    int src = n2 * tp1 + tau2;
    o1 = a1s[src]; o2 = a2s[src];
  }
  A1[(bt << 9) + l] = o1;
  A2[(bt << 9) + l] = o2;

  float nvv = nv[(b << 9) + l];
  float v1 = block_reduce_sum(o1 * nvv, red);
  float v2 = block_reduce_sum(o2 * nvv, red);
  if (l == 0) { V1[bt] = v1; V2[bt] = v2; }

  // op logits: (opq row) @ Wsel  (512x9 GEMV via 9 block reductions)
  float q = OPQ[(bt << 9) + l];
  float lg[NOPS9];
#pragma unroll
  for (int j = 0; j < NOPS9; ++j)
    lg[j] = block_reduce_sum(q * Wsel[l * NOPS9 + j], red);

  if (l == 0) {
    int am = 0; float best = -__builtin_inff();
    for (int j = 0; j < NOPS9; ++j) {
      float z = fminf(fmaxf(lg[j] + bsel[j], -40.f), 40.f);
      float yv = z + gumbel_noise(3u, b, t, (unsigned)j);
      if (yv > best) { best = yv; am = j; }
    }
    const float eps = 1e-8f;
    float add = v1 + v2, ident = v1, mul = v1 * v2, sub = v1 - v2;
    float dv = v1 / (fabsf(v2) + eps);
    dv = fminf(fmaxf(dv, -1e6f), 1e6f);
    float xs = fminf(fmaxf(fabsf(v1) + eps, 1e-8f), 1e3f);
    float ys = fminf(fmaxf(v2, -6.f), 6.f);
    float sg = (v1 > 0.f) ? 1.f : ((v1 < 0.f) ? -1.f : 0.f);
    float pw = __powf(xs, ys) * sg;
    pw = fminf(fmaxf(pw, -1e6f), 1e6f);
    float lgo = fminf(fmaxf(__logf(fabsf(v1) + eps), -20.f), 20.f);
    float mx = fmaxf(v1, v2), mn = fminf(v1, v2);
    float outs[NOPS9] = {add, ident, mul, sub, dv, pw, lgo, mx, mn};
    float nvnew = 0.f;
    for (int j = 0; j < NOPS9; ++j) {
      float w = ((j == am) ? 1.f : 0.f) + 1e-10f;   // hard gumbel forward
      OPW[bt * NOPS9 + j] = w;
      nvnew += w * outs[j];
    }
    outv[((b * (NN + 1) + NN) * TT) + t] = nvnew;
  }
}

// ---------------------------------------------------------------------------
// Epilogue: new_emb = gemm + 0.5(e1+e2) + v1*wse3 + v2*wse4 + cc + opw@Wmlp[2048:]
// writes directly into output slot [b, N, t, :]
// ---------------------------------------------------------------------------
__global__ void epilogue_kernel(const float* __restrict__ GMLP,
                                const float* __restrict__ E12,
                                const float* __restrict__ V1,
                                const float* __restrict__ V2,
                                const float* __restrict__ OPW,
                                const float* __restrict__ Wmlp,
                                const float* __restrict__ wse3,
                                const float* __restrict__ wse4,
                                const float* __restrict__ cc,
                                float* __restrict__ out) {
  int i = blockIdx.x * blockDim.x + threadIdx.x;
  if (i >= BB * TT * HH) return;
  int bt = i >> 9, h = i & 511;
  int b = bt >> 6, t = bt & 63;
  float e1 = E12[bt * 1024 + h];
  float e2 = E12[bt * 1024 + 512 + h];
  float v = GMLP[i] + 0.5f * (e1 + e2) + V1[bt] * wse3[h] + V2[bt] * wse4[h] + cc[h];
#pragma unroll
  for (int j = 0; j < NOPS9; ++j)
    v += OPW[bt * NOPS9 + j] * Wmlp[(2048 + j) * HH + h];
  out[(((size_t)(b * (NN + 1) + NN) * TT + t) << 9) + h] = v;
}

// ---------------------------------------------------------------------------
extern "C" void kernel_launch(void* const* d_in, const int* in_sizes, int n_in,
                              void* d_out, int out_size, void* d_ws, size_t ws_size,
                              hipStream_t stream) {
  const float* node_embeds = (const float*)d_in[0];
  const float* node_values = (const float*)d_in[1];
  const float* operand_ctx = (const float*)d_in[2];
  const float* op_ctx      = (const float*)d_in[3];
  const float* Wq1  = (const float*)d_in[4];
  const float* bq1  = (const float*)d_in[5];
  const float* Wq2  = (const float*)d_in[6];
  const float* bq2  = (const float*)d_in[7];
  const float* Wopq = (const float*)d_in[8];
  const float* bopq = (const float*)d_in[9];
  const float* Wk   = (const float*)d_in[10];
  const float* bk   = (const float*)d_in[11];
  const float* Wsel = (const float*)d_in[12];
  const float* bsel = (const float*)d_in[13];
  const float* Wse  = (const float*)d_in[14];
  const float* bse  = (const float*)d_in[15];
  const float* Wmlp = (const float*)d_in[16];
  const float* bmlp = (const float*)d_in[17];
  const float* step0 = (const float*)d_in[18];
  float* out = (float*)d_out;
  float* ws  = (float*)d_ws;

  const int BT = BB * TT;               // 256
  const int BTH = BT * HH;              // 131072
  size_t o = 0;
  float* OPND = ws + o; o += BTH;
  float* OPC  = ws + o; o += BTH;
  float* KALL = ws + o; o += (size_t)BB * NTOT * HH;   // 1048576
  float* Q1   = ws + o; o += BTH;
  float* Q2   = ws + o; o += BTH;
  float* OPQ  = ws + o; o += BTH;
  float* S1   = ws + o; o += BTH;
  float* S2   = ws + o; o += BTH;
  float* A1   = ws + o; o += BTH;
  float* A2   = ws + o; o += BTH;
  float* V1   = ws + o; o += BT;
  float* V2   = ws + o; o += BT;
  float* OPW  = ws + o; o += BT * NOPS9;
  float* E12  = ws + o; o += (size_t)BT * 1024;
  float* GMLP = ws + o; o += BTH;
  float* WSE3 = ws + o; o += HH;
  float* WSE4 = ws + o; o += HH;
  float* CC   = ws + o; o += HH;

  // independent preps
  addstep_kernel<<<(BTH + 255) / 256, 256, 0, stream>>>(operand_ctx, op_ctx, step0, OPND, OPC);
  prep_kernel<<<2, 256, 0, stream>>>(Wse, bse, Wmlp, bmlp, WSE3, WSE4, CC);
  {
    int tot = BB * NN * TT * HH + BB * NN * TT;
    copy_kernel<<<(tot + 255) / 256, 256, 0, stream>>>(node_embeds, node_values, out);
  }

  const int WPB = 4;  // waves per block (blockDim 128)
  // 1) K = node_embeds @ Wk + bk   (M=2048, N=512, K=512)
  {
    int tiles = (2048 >> 5) * (512 >> 6);
    wmma_gemm_kernel<0><<<dim3((tiles + WPB - 1) / WPB, 1), 128, 0, stream>>>(
        node_embeds, Wk, bk, KALL, 2048, 512, 512, 512, 512, 0, 0, 0);
  }
  // 2) q1/q2/opq  (M=256, N=512, K=512)
  {
    int tiles = (256 >> 5) * (512 >> 6);
    dim3 g((tiles + WPB - 1) / WPB, 1);
    wmma_gemm_kernel<0><<<g, 128, 0, stream>>>(OPND, Wq1, bq1, Q1, 256, 512, 512, 512, 512, 0, 0, 0);
    wmma_gemm_kernel<0><<<g, 128, 0, stream>>>(OPND, Wq2, bq2, Q2, 256, 512, 512, 512, 512, 0, 0, 0);
    wmma_gemm_kernel<0><<<g, 128, 0, stream>>>(OPC, Wopq, bopq, OPQ, 256, 512, 512, 512, 512, 0, 0, 0);
  }
  // 3) scores: per b, S = q @ K^T  (M=64, N=512(nt), K=512(h), transB)
  {
    int tiles = (64 >> 5) * (512 >> 6);
    dim3 g((tiles + WPB - 1) / WPB, BB);
    wmma_gemm_kernel<1><<<g, 128, 0, stream>>>(Q1, KALL, nullptr, S1, 64, 512, 512,
        512, 512, (long)TT * HH, (long)NTOT * HH, (long)TT * NTOT);
    wmma_gemm_kernel<1><<<g, 128, 0, stream>>>(Q2, KALL, nullptr, S2, 64, 512, 512,
        512, 512, (long)TT * HH, (long)NTOT * HH, (long)TT * NTOT);
  }
  // 4) softmax/scatter/op-select (writes new_values slot of d_out directly)
  attn_kernel<<<BT, 512, 0, stream>>>(S1, S2, OPQ, Wsel, bsel, node_values,
      A1, A2, V1, V2, OPW, out + (size_t)BB * (NN + 1) * TT * HH);
  // 5) e1/e2: per b, e = A_scat @ node_embeds[b]  (M=64, N=512, K=512)
  {
    int tiles = (64 >> 5) * (512 >> 6);
    dim3 g((tiles + WPB - 1) / WPB, BB);
    wmma_gemm_kernel<0><<<g, 128, 0, stream>>>(A1, node_embeds, nullptr, E12, 64, 512, 512,
        512, 1024, (long)TT * NTOT, (long)NTOT * HH, (long)TT * 1024);
    wmma_gemm_kernel<0><<<g, 128, 0, stream>>>(A2, node_embeds, nullptr, E12 + 512, 64, 512, 512,
        512, 1024, (long)TT * NTOT, (long)NTOT * HH, (long)TT * 1024);
  }
  // 6) MLP main GEMM: [e1|e2] (256x1024) @ Wmlp[0:1024] (1024x512)
  {
    int tiles = (256 >> 5) * (512 >> 6);
    wmma_gemm_kernel<0><<<dim3((tiles + WPB - 1) / WPB, 1), 128, 0, stream>>>(
        E12, Wmlp, nullptr, GMLP, 256, 512, 1024, 1024, 512, 0, 0, 0);
  }
  // 7) epilogue -> new_emb slot of d_out
  epilogue_kernel<<<(BTH + 255) / 256, 256, 0, stream>>>(
      GMLP, E12, V1, V2, OPW, Wmlp, WSE3, WSE4, CC, out);

  (void)in_sizes; (void)n_in; (void)out_size; (void)ws_size;
}
